// MultiHeadSelfAttention_77180562309123
// MI455X (gfx1250) — matrix operand order repaired and checked by reading, not compiled, not run on hardware
//
#include <hip/hip_runtime.h>

#define BDIM 4
#define SDIM 2048
#define DDIM 1024
#define HDIM 16
#define EDIM 64
#define HE   (HDIM * EDIM)   // 1024
#define BS   (BDIM * SDIM)   // 8192

typedef __bf16 bf16;
typedef __attribute__((ext_vector_type(16))) __bf16 v16bf;
typedef __attribute__((ext_vector_type(8)))  __bf16 v8bf;
typedef __attribute__((ext_vector_type(8)))  float  v8f;

#define WMMA_BF16(a, b, c) \
  __builtin_amdgcn_wmma_f32_16x16x32_bf16(false, (a), false, (b), (short)0, (c), false, false)

// ---------------------------------------------------------------------------
// Fragment loaders (wave32 layouts per CDNA5 ISA 7.12.2)
// A (16x32 bf16): lane m(=lane&15) holds K chunks [kh*8, kh*8+8) and
//                 [16+kh*8, 16+kh*8+8) of row m, kh = lane>>4.
// B (32x16 bf16): lane n(=lane&15) holds K chunks [kh*8, kh*8+8) and
//                 [16+kh*8, 16+kh*8+8) of column n, kh = lane>>4.
// ---------------------------------------------------------------------------
static __device__ __forceinline__ v16bf load_a_frag(const bf16* __restrict__ base,
                                                    int ld, int kbase, int lane) {
  const int row = lane & 15, kh = lane >> 4;
  union { v16bf v; v8bf h[2]; } u;
  const bf16* r = base + (size_t)row * ld + kbase;
  u.h[0] = *(const v8bf*)(r + kh * 8);
  u.h[1] = *(const v8bf*)(r + 16 + kh * 8);
  return u.v;
}

static __device__ __forceinline__ v16bf load_b_frag(const bf16* __restrict__ base,
                                                    int ld, int kbase, int lane) {
  const int col = lane & 15, kh = lane >> 4;
  union { v16bf v; v8bf h[2]; } u;
  const bf16* r = base + (size_t)col * ld + kbase;
  u.h[0] = *(const v8bf*)(r + kh * 8);
  u.h[1] = *(const v8bf*)(r + 16 + kh * 8);
  return u.v;
}

// ---------------------------------------------------------------------------
// CDNA5 async global->LDS copy (ASYNCcnt family, ISA 08_async_tensor 4.2).
// vdst VGPR = per-lane LDS byte address, vaddr = per-lane 64-bit global addr.
// ---------------------------------------------------------------------------
static __device__ __forceinline__ void async_b128(unsigned int lds_off, const void* gaddr) {
  asm volatile("global_load_async_to_lds_b128 %0, %1, off"
               :: "v"(lds_off), "v"(gaddr) : "memory");
}
#define WAIT_ASYNC_16() asm volatile("s_wait_asynccnt 0x10" ::: "memory")
#define WAIT_ASYNC_0()  asm volatile("s_wait_asynccnt 0x0"  ::: "memory")
#define WAIT_DS_0()     asm volatile("s_wait_dscnt 0x0"     ::: "memory")

// ---------------------------------------------------------------------------
// Prep kernels: fp32 -> bf16 convert, weight transposes
// ---------------------------------------------------------------------------
__global__ void mhsa_cvt_x(const float* __restrict__ in, bf16* __restrict__ out, int n) {
  int i = blockIdx.x * 256 + threadIdx.x;
  if (i < n) out[i] = (bf16)in[i];
}

// Wq[h, d, e] -> wt[h, e, d]
__global__ void mhsa_tr_w(const float* __restrict__ w, bf16* __restrict__ wt) {
  int i = blockIdx.x * 256 + threadIdx.x;
  if (i < HDIM * DDIM * EDIM) {
    int hh = i / (DDIM * EDIM);
    int r  = i - hh * (DDIM * EDIM);
    int d  = r / EDIM;
    int e  = r - d * EDIM;
    wt[((size_t)hh * EDIM + e) * DDIM + d] = (bf16)w[i];
  }
}

// Wo[c, d] -> wt[d, c]
__global__ void mhsa_tr_wo(const float* __restrict__ w, bf16* __restrict__ wt) {
  int i = blockIdx.x * 256 + threadIdx.x;
  if (i < HE * DDIM) {
    int c = i / DDIM;
    int d = i - c * DDIM;
    wt[(size_t)d * HE + c] = (bf16)w[i];
  }
}

// ---------------------------------------------------------------------------
// QKV projection, register-pipelined: load k+32 fragments while the k-step
// WMMAs execute, so the matrix pipe is not serialized on s_wait_loadcnt 0.
// V is stored transposed [b,h,e,s] so attention's B-operand is contiguous.
// ---------------------------------------------------------------------------
__global__ void mhsa_qkv(const bf16* __restrict__ xb,
                         const bf16* __restrict__ wqt, const bf16* __restrict__ wkt,
                         const bf16* __restrict__ wvt,
                         bf16* __restrict__ Qb, bf16* __restrict__ Kb,
                         bf16* __restrict__ Vtb) {
  const int lane = threadIdx.x;
  const int q0   = blockIdx.x * 16;
  const int h    = blockIdx.y / 3;
  const int mat  = blockIdx.y % 3;
  const int b    = blockIdx.z;

  const bf16* A  = xb + ((size_t)b * SDIM + q0) * DDIM;
  const bf16* Wt = (mat == 0 ? wqt : (mat == 1 ? wkt : wvt)) + (size_t)h * EDIM * DDIM;

  v8f acc[4] = {};
  v16bf a = load_a_frag(A, DDIM, 0, lane);
  v16bf bb[4];
#pragma unroll
  for (int n = 0; n < 4; ++n) bb[n] = load_b_frag(Wt + (size_t)(n * 16) * DDIM, DDIM, 0, lane);

  for (int k = 0; k < DDIM; k += 32) {
    const v16bf ac = a;
    v16bf bc[4];
#pragma unroll
    for (int n = 0; n < 4; ++n) bc[n] = bb[n];
    const int k2 = k + 32;
    if (k2 < DDIM) {
      __builtin_prefetch(A + (size_t)(lane & 15) * DDIM + k2 + 64, 0, 3);
      a = load_a_frag(A, DDIM, k2, lane);
#pragma unroll
      for (int n = 0; n < 4; ++n)
        bb[n] = load_b_frag(Wt + (size_t)(n * 16) * DDIM, DDIM, k2, lane);
    }
#pragma unroll
    for (int n = 0; n < 4; ++n) acc[n] = WMMA_BF16(ac, bc[n], acc[n]);
  }

  const int hi = lane >> 4, cn = lane & 15;
  if (mat < 2) {
    bf16* out = (mat == 0 ? Qb : Kb) + ((size_t)(b * HDIM + h) * SDIM + q0) * EDIM;
#pragma unroll
    for (int n = 0; n < 4; ++n)
#pragma unroll
      for (int j = 0; j < 8; ++j) {
        const int m = hi * 8 + j;
        out[(size_t)m * EDIM + n * 16 + cn] = (bf16)acc[n][j];
      }
  } else {
    bf16* out = Vtb + (size_t)(b * HDIM + h) * EDIM * SDIM;
#pragma unroll
    for (int n = 0; n < 4; ++n)
#pragma unroll
      for (int j = 0; j < 8; ++j) {
        const int m = hi * 8 + j;
        out[(size_t)(n * 16 + cn) * SDIM + q0 + m] = (bf16)acc[n][j];
      }
  }
}

// ---------------------------------------------------------------------------
// Causal flash attention with double-buffered async K/V staging in LDS.
//   - K tile (32 x 64 bf16 = 4KB, contiguous) and V tile (64 rows x 64B from
//     Vt) are pulled with global_load_async_to_lds_b128 (8 instrs each).
//   - Next t-block's copies are issued before computing the current block;
//     s_wait_asynccnt 16 fences exactly one in-flight group (in-order cnt).
//   - P (softmax probs) is transposed C/D->A layout via a 1KB LDS bounce
//     fenced by s_wait_dscnt.
// ---------------------------------------------------------------------------
static __device__ __forceinline__ void stage_k(const bf16* __restrict__ K, int t0,
                                               const bf16* dst, int lane) {
  const char* src = (const char*)(K + (size_t)t0 * EDIM);
  const unsigned int d = (unsigned int)(uintptr_t)dst;
#pragma unroll
  for (int c = 0; c < 8; ++c)
    async_b128(d + c * 512 + lane * 16, src + c * 512 + lane * 16);
}

static __device__ __forceinline__ void stage_v(const bf16* __restrict__ Vt, int t0,
                                               const bf16* dst, int lane) {
  const unsigned int d = (unsigned int)(uintptr_t)dst;
  const int r0 = lane >> 2, ch = lane & 3;   // 8 rows x 4 chunks per instr
#pragma unroll
  for (int c = 0; c < 8; ++c) {
    const int e = c * 8 + r0;
    const char* src = (const char*)(Vt + (size_t)e * SDIM + t0) + ch * 16;
    async_b128(d + e * 64 + ch * 16, src);
  }
}

__global__ void mhsa_attn(const bf16* __restrict__ Qb, const bf16* __restrict__ Kb,
                          const bf16* __restrict__ Vtb, bf16* __restrict__ headsb) {
  __shared__ __align__(16) bf16 kbuf[2][32 * EDIM];  // [t-local 0..31][e 0..63]
  __shared__ __align__(16) bf16 vbuf[2][EDIM * 32];  // [e 0..63][t-local 0..31]
  __shared__ __align__(16) bf16 pbuf[16 * 32];

  const int lane = threadIdx.x;
  const int q0   = blockIdx.x * 16;
  const int h    = blockIdx.y;
  const int b    = blockIdx.z;
  const size_t bh = (size_t)(b * HDIM + h);

  const bf16* Q  = Qb  + (bh * SDIM + q0) * EDIM;
  const bf16* K  = Kb  + bh * SDIM * EDIM;
  const bf16* Vt = Vtb + bh * EDIM * SDIM;
  const int hi = lane >> 4, cn = lane & 15;
  const int tend = q0 + 15;

  // Q fragments (K-dim = E = 64 -> two K=32 steps), reused for all t.
  const v16bf qf0 = load_a_frag(Q, EDIM, 0, lane);
  const v16bf qf1 = load_a_frag(Q, EDIM, 32, lane);

  float mrow[8], lrow[8];
  v8f o[4] = {};
#pragma unroll
  for (int j = 0; j < 8; ++j) { mrow[j] = -1e30f; lrow[j] = 0.f; }

  // Prologue: stage first K/V block into buffer 0.
  stage_k(K, 0, kbuf[0], lane);
  stage_v(Vt, 0, vbuf[0], lane);

  int buf = 0;
  for (int t0 = 0; t0 <= tend; t0 += 32) {
    const bool more = (t0 + 32 <= tend);
    if (more) {
      stage_k(K, t0 + 32, kbuf[buf ^ 1], lane);
      stage_v(Vt, t0 + 32, vbuf[buf ^ 1], lane);
      WAIT_ASYNC_16();          // current block's 16 copies complete
    } else {
      WAIT_ASYNC_0();
    }
    const bf16* kb = kbuf[buf];
    const bf16* vb = vbuf[buf];

    v8f s0 = {}, s1 = {};
    {
      v16bf kf = load_b_frag(kb, EDIM, 0, lane);                 // t-tile 0, e 0..31
      s0 = WMMA_BF16(qf0, kf, s0);
      kf = load_b_frag(kb, EDIM, 32, lane);                      // t-tile 0, e 32..63
      s0 = WMMA_BF16(qf1, kf, s0);
      kf = load_b_frag(kb + 16 * EDIM, EDIM, 0, lane);           // t-tile 1
      s1 = WMMA_BF16(qf0, kf, s1);
      kf = load_b_frag(kb + 16 * EDIM, EDIM, 32, lane);
      s1 = WMMA_BF16(qf1, kf, s1);
    }

    float alpha[8];
#pragma unroll
    for (int j = 0; j < 8; ++j) {
      const int qa = q0 + hi * 8 + j;
      float v0 = s0[j] * 0.125f; if (t0 + cn > qa)      v0 = -1e30f;  // causal mask
      float v1 = s1[j] * 0.125f; if (t0 + 16 + cn > qa) v1 = -1e30f;
      float tm = fmaxf(v0, v1);
#pragma unroll
      for (int d2 = 1; d2 < 16; d2 <<= 1) tm = fmaxf(tm, __shfl_xor(tm, d2, 32));
      const float mn = fmaxf(mrow[j], tm);
      alpha[j] = __expf(mrow[j] - mn);
      mrow[j]  = mn;
      const float p0 = __expf(v0 - mn), p1 = __expf(v1 - mn);
      float rs = p0 + p1;
#pragma unroll
      for (int d2 = 1; d2 < 16; d2 <<= 1) rs += __shfl_xor(rs, d2, 32);
      lrow[j] = lrow[j] * alpha[j] + rs;
      const int m = hi * 8 + j;
      pbuf[m * 32 + cn]      = (bf16)p0;   // stage P for layout transpose
      pbuf[m * 32 + 16 + cn] = (bf16)p1;
    }
#pragma unroll
    for (int n = 0; n < 4; ++n)
#pragma unroll
      for (int j = 0; j < 8; ++j) o[n][j] *= alpha[j];

    WAIT_DS_0();                                       // P stores -> P reads
    const v16bf pf = load_a_frag((const bf16*)pbuf, 32, 0, lane);
#pragma unroll
    for (int n = 0; n < 4; ++n) {
      v16bf vf = load_b_frag(vb + (size_t)(n * 16) * 32, 32, 0, lane);
      o[n] = WMMA_BF16(pf, vf, o[n]);
    }
    buf ^= 1;
  }

  // heads layout: [b, s, h*E + e]
  bf16* Ho = headsb + ((size_t)b * SDIM + q0) * HE + h * EDIM;
#pragma unroll
  for (int n = 0; n < 4; ++n)
#pragma unroll
    for (int j = 0; j < 8; ++j) {
      const int m = hi * 8 + j;
      Ho[(size_t)m * HE + n * 16 + cn] = (bf16)(o[n][j] / lrow[j]);
    }
}

// ---------------------------------------------------------------------------
// Output projection + residual: y = x + heads @ Wo (fp32), register-pipelined.
// ---------------------------------------------------------------------------
__global__ void mhsa_proj(const bf16* __restrict__ headsb, const bf16* __restrict__ wot,
                          const float* __restrict__ x, float* __restrict__ yres) {
  const int lane = threadIdx.x;
  const size_t m0 = (size_t)blockIdx.x * 16;
  const int    n0 = blockIdx.y * 16;
  const bf16* A  = headsb + m0 * HE;
  const bf16* Bt = wot + (size_t)n0 * HE;

  v8f acc = {};
  v16bf a  = load_a_frag(A, HE, 0, lane);
  v16bf bb = load_b_frag(Bt, HE, 0, lane);
  for (int k = 0; k < HE; k += 32) {
    const v16bf ac = a, bc = bb;
    const int k2 = k + 32;
    if (k2 < HE) {
      __builtin_prefetch(A + (size_t)(lane & 15) * HE + k2 + 64, 0, 3);
      a  = load_a_frag(A, HE, k2, lane);
      bb = load_b_frag(Bt, HE, k2, lane);
    }
    acc = WMMA_BF16(ac, bc, acc);
  }

  const int hi = lane >> 4, cn = lane & 15;
#pragma unroll
  for (int j = 0; j < 8; ++j) {
    const int m = hi * 8 + j;
    const size_t idx = (m0 + m) * DDIM + n0 + cn;
    yres[idx] = x[idx] + acc[j];
  }
}

// ---------------------------------------------------------------------------
// LayerNorm (axis=-1, eps=1e-3), one 256-thread block per row of 1024.
// ---------------------------------------------------------------------------
__global__ void mhsa_ln(const float* __restrict__ yres, const float* __restrict__ gamma,
                        const float* __restrict__ beta, float* __restrict__ out) {
  const int tid = threadIdx.x;
  const float* row = yres + (size_t)blockIdx.x * DDIM;

  float s = 0.f;
  for (int i = tid; i < DDIM; i += 256) s += row[i];
#pragma unroll
  for (int d2 = 1; d2 < 32; d2 <<= 1) s += __shfl_xor(s, d2, 32);
  __shared__ float red[8];
  if ((tid & 31) == 0) red[tid >> 5] = s;
  __syncthreads();
  float tot = 0.f;
#pragma unroll
  for (int w = 0; w < 8; ++w) tot += red[w];
  const float mu = tot * (1.0f / DDIM);

  float vs = 0.f;
  for (int i = tid; i < DDIM; i += 256) { const float d = row[i] - mu; vs += d * d; }
#pragma unroll
  for (int d2 = 1; d2 < 32; d2 <<= 1) vs += __shfl_xor(vs, d2, 32);
  __shared__ float red2[8];
  if ((tid & 31) == 0) red2[tid >> 5] = vs;
  __syncthreads();
  float tv = 0.f;
#pragma unroll
  for (int w = 0; w < 8; ++w) tv += red2[w];
  const float inv = rsqrtf(tv * (1.0f / DDIM) + 1e-3f);

  float* o = out + (size_t)blockIdx.x * DDIM;
  for (int i = tid; i < DDIM; i += 256)
    o[i] = (row[i] - mu) * inv * gamma[i] + beta[i];
}

// ---------------------------------------------------------------------------
// Launcher
// ---------------------------------------------------------------------------
extern "C" void kernel_launch(void* const* d_in, const int* in_sizes, int n_in,
                              void* d_out, int out_size, void* d_ws, size_t ws_size,
                              hipStream_t stream) {
  (void)in_sizes; (void)n_in; (void)out_size; (void)ws_size;
  const float* x     = (const float*)d_in[0];
  const float* Wq    = (const float*)d_in[1];
  const float* Wk    = (const float*)d_in[2];
  const float* Wv    = (const float*)d_in[3];
  const float* Wo    = (const float*)d_in[4];
  const float* gamma = (const float*)d_in[5];
  const float* beta  = (const float*)d_in[6];
  float* out = (float*)d_out;

  char* ws = (char*)d_ws;
  size_t off = 0;
  auto alloc = [&](size_t bytes) -> void* {
    void* p = ws + off;
    off += (bytes + 255) & ~(size_t)255;
    return p;
  };
  bf16* xb     = (bf16*)alloc((size_t)BS * DDIM * 2);            // x in bf16
  bf16* wqt    = (bf16*)alloc((size_t)HDIM * EDIM * DDIM * 2);   // Wq^T [h,e,d]
  bf16* wkt    = (bf16*)alloc((size_t)HDIM * EDIM * DDIM * 2);
  bf16* wvt    = (bf16*)alloc((size_t)HDIM * EDIM * DDIM * 2);
  bf16* wot    = (bf16*)alloc((size_t)DDIM * HE * 2);            // Wo^T [d,c]
  bf16* Qb     = (bf16*)alloc((size_t)BDIM * HDIM * SDIM * EDIM * 2);
  bf16* Kb     = (bf16*)alloc((size_t)BDIM * HDIM * SDIM * EDIM * 2);
  bf16* Vtb    = (bf16*)alloc((size_t)BDIM * HDIM * EDIM * SDIM * 2);  // [b,h,e,s]
  bf16* headsb = (bf16*)alloc((size_t)BS * HE * 2);
  float* yres  = (float*)alloc((size_t)BS * DDIM * 4);

  const int n1 = BS * DDIM;
  mhsa_cvt_x<<<(n1 + 255) / 256, 256, 0, stream>>>(x, xb, n1);
  const int n2 = HDIM * DDIM * EDIM;
  mhsa_tr_w<<<(n2 + 255) / 256, 256, 0, stream>>>(Wq, wqt);
  mhsa_tr_w<<<(n2 + 255) / 256, 256, 0, stream>>>(Wk, wkt);
  mhsa_tr_w<<<(n2 + 255) / 256, 256, 0, stream>>>(Wv, wvt);
  const int n3 = HE * DDIM;
  mhsa_tr_wo<<<(n3 + 255) / 256, 256, 0, stream>>>(Wo, wot);

  mhsa_qkv<<<dim3(SDIM / 16, HDIM * 3, BDIM), 32, 0, stream>>>(
      xb, wqt, wkt, wvt, Qb, Kb, Vtb);

  mhsa_attn<<<dim3(SDIM / 16, HDIM, BDIM), 32, 0, stream>>>(Qb, Kb, Vtb, headsb);

  mhsa_proj<<<dim3(BS / 16, DDIM / 16), 32, 0, stream>>>(headsb, wot, x, yres);

  mhsa_ln<<<BS, 256, 0, stream>>>(yres, gamma, beta, out);
}
